// MultiHeadAttention_21500606284356
// MI455X (gfx1250) — compile-verified, matrix operand order checked by reading, not run on hardware
//
#include <hip/hip_runtime.h>
#include <hip/hip_bf16.h>
#include <stdint.h>

// Problem constants (from reference)
#define BATCH   2
#define SEQ     2048
#define DMODEL  1024
#define NHEADS  16
#define DKH     64

typedef __attribute__((ext_vector_type(16))) __bf16 v16bf;
typedef __attribute__((ext_vector_type(8)))  float  v8f;

struct __align__(16) U128 { uint32_t x[4]; };

union Frag16 {
    v16bf v;
    U128  q[2];
    unsigned short u[16];
};

__device__ __forceinline__ v8f vzero8() {
    v8f z = {0.f, 0.f, 0.f, 0.f, 0.f, 0.f, 0.f, 0.f};
    return z;
}

__device__ __forceinline__ unsigned short f2bf(float f) {
    union { float f; uint32_t u; } c; c.f = f;
    uint32_t u = c.u;
    u += 0x7FFFu + ((u >> 16) & 1u);   // round-to-nearest-even
    return (unsigned short)(u >> 16);
}

__device__ __forceinline__ uint32_t pack2bf(float lo, float hi) {
    return (uint32_t)f2bf(lo) | ((uint32_t)f2bf(hi) << 16);
}

// Load a 16(M/N-rows) x 32(K) bf16 operand fragment from a row-major array.
// Layout per CDNA5 ISA:
//   lane = (row & 15) + 16*half ; VGPR v slot s holds K = (v>>2)*16 + half*8 + (v&3)*2 + s
// => per lane: two contiguous 8-element (16B) runs at K = half*8 and 16+half*8.
__device__ __forceinline__ Frag16 load_frag(const unsigned short* base, int ld, int lane) {
    const int r  = lane & 15;
    const int hf = lane >> 4;
    const unsigned short* p = base + (size_t)r * ld + hf * 8;
    Frag16 f;
    f.q[0] = *reinterpret_cast<const U128*>(p);
    f.q[1] = *reinterpret_cast<const U128*>(p + 16);
    return f;
}

__device__ __forceinline__ v8f wmma_bf16(const Frag16& a, const Frag16& b, v8f c) {
    return __builtin_amdgcn_wmma_f32_16x16x32_bf16(
        /*neg_a=*/false, a.v, /*neg_b=*/false, b.v,
        /*c_mod=*/(short)0, c, /*reuse_a=*/false, /*reuse_b=*/false);
}

// ---------------------------------------------------------------------------
// GEMM with bias:  Out = alpha * (A[M,K] @ W[N,K]^T + bias[N])
// Tile: 128(M) x 128(N), K-step 64. 256 threads = 8 waves; wave = 32x64 subtile.
// MODE 0: f32 output, row-major [M, DMODEL]
// MODE 1: bf16 output, permuted  [B][H][S][dk]   (for Q, K)
// MODE 2: bf16 output, transposed [B][H][dk][S]  (for V)
// A_BF16: A is bf16 (ctx) instead of f32.
// ---------------------------------------------------------------------------
#define KSTEP 64
#define LDSS  72   // padded LDS row stride (bf16): 144B rows -> 36-bank step, conflict-free

template <int MODE, bool A_BF16>
__global__ __launch_bounds__(256, 2)
void gemm_bias_wmma(const void* __restrict__ Ain,
                    const float* __restrict__ W,
                    const float* __restrict__ bias,
                    void* __restrict__ Out,
                    float alpha) {
    constexpr int K = DMODEL;
    __shared__ __align__(16) unsigned short As[128 * LDSS];
    __shared__ __align__(16) unsigned short Bs[128 * LDSS];

    const int tid  = threadIdx.x;
    const int lane = tid & 31;
    const int wv   = tid >> 5;
    const int wm   = wv & 3;           // 4 M-subtiles of 32
    const int wn   = wv >> 2;          // 2 N-subtiles of 64
    const int n0   = blockIdx.x * 128;
    const int m0   = blockIdx.y * 128;

    const int lrow = tid >> 1;         // 0..127 staging row
    const int lcol = (tid & 1) * 32;   // 0 or 32 (32 elems per thread per tile)

    v8f acc[2][4];
#pragma unroll
    for (int i = 0; i < 2; ++i)
#pragma unroll
        for (int j = 0; j < 4; ++j) acc[i][j] = vzero8();

    for (int k0 = 0; k0 < K; k0 += KSTEP) {
        // --- stage A tile (128x64) to LDS as bf16 ---
        if (A_BF16) {
            const unsigned short* src =
                (const unsigned short*)Ain + (size_t)(m0 + lrow) * K + k0 + lcol;
#pragma unroll
            for (int c = 0; c < 4; ++c) {
                U128 a = *reinterpret_cast<const U128*>(src + c * 8);
                *reinterpret_cast<U128*>(&As[lrow * LDSS + lcol + c * 8]) = a;
            }
        } else {
            const float* src = (const float*)Ain + (size_t)(m0 + lrow) * K + k0 + lcol;
#pragma unroll
            for (int c = 0; c < 4; ++c) {
                float4 x0 = *reinterpret_cast<const float4*>(src + c * 8);
                float4 x1 = *reinterpret_cast<const float4*>(src + c * 8 + 4);
                U128 p;
                p.x[0] = pack2bf(x0.x, x0.y); p.x[1] = pack2bf(x0.z, x0.w);
                p.x[2] = pack2bf(x1.x, x1.y); p.x[3] = pack2bf(x1.z, x1.w);
                *reinterpret_cast<U128*>(&As[lrow * LDSS + lcol + c * 8]) = p;
            }
        }
        // --- stage B tile: rows of W (output features) ---
        {
            const float* src = W + (size_t)(n0 + lrow) * K + k0 + lcol;
#pragma unroll
            for (int c = 0; c < 4; ++c) {
                float4 x0 = *reinterpret_cast<const float4*>(src + c * 8);
                float4 x1 = *reinterpret_cast<const float4*>(src + c * 8 + 4);
                U128 p;
                p.x[0] = pack2bf(x0.x, x0.y); p.x[1] = pack2bf(x0.z, x0.w);
                p.x[2] = pack2bf(x1.x, x1.y); p.x[3] = pack2bf(x1.z, x1.w);
                *reinterpret_cast<U128*>(&Bs[lrow * LDSS + lcol + c * 8]) = p;
            }
        }
        __syncthreads();

#pragma unroll
        for (int kk = 0; kk < 2; ++kk) {
            Frag16 af[2], bfr[4];
#pragma unroll
            for (int mi = 0; mi < 2; ++mi)
                af[mi] = load_frag(&As[(wm * 32 + mi * 16) * LDSS + kk * 32], LDSS, lane);
#pragma unroll
            for (int ni = 0; ni < 4; ++ni)
                bfr[ni] = load_frag(&Bs[(wn * 64 + ni * 16) * LDSS + kk * 32], LDSS, lane);
#pragma unroll
            for (int mi = 0; mi < 2; ++mi)
#pragma unroll
                for (int ni = 0; ni < 4; ++ni)
                    acc[mi][ni] = wmma_bf16(af[mi], bfr[ni], acc[mi][ni]);
        }
        __syncthreads();
    }

    // --- epilogue: bias, scale, scatter store ---
    const int nb = n0 + wn * 64;
    const int mb = m0 + wm * 32;
    const int hf = lane >> 4;
    const int lc = lane & 15;
#pragma unroll
    for (int mi = 0; mi < 2; ++mi) {
#pragma unroll
        for (int ni = 0; ni < 4; ++ni) {
            const int gcol = nb + ni * 16 + lc;
            const float bv = bias[gcol];
#pragma unroll
            for (int r = 0; r < 8; ++r) {
                const int grow = mb + mi * 16 + hf * 8 + r;   // C-layout row
                const float val = (acc[mi][ni][r] + bv) * alpha;
                if (MODE == 0) {
                    ((float*)Out)[(size_t)grow * DMODEL + gcol] = val;
                } else {
                    const int b = grow >> 11, s = grow & (SEQ - 1);
                    const int h = gcol >> 6, d = gcol & 63;
                    size_t addr;
                    if (MODE == 1)
                        addr = (((size_t)b * NHEADS + h) * SEQ + s) * DKH + d;
                    else
                        addr = (((size_t)b * NHEADS + h) * DKH + d) * SEQ + s;
                    ((unsigned short*)Out)[addr] = f2bf(val);
                }
            }
        }
    }
}

// ---------------------------------------------------------------------------
// Flash attention: grid (SEQ/128, BATCH*NHEADS), 256 threads = 8 waves.
// Each wave handles 16 query rows; kv loop in tiles of 64.
// Q/K: bf16 [B][H][S][dk]; Q pre-scaled by log2(e)/sqrt(dk) => base-2 softmax.
// V: bf16 [B][H][dk][S].  Output ctx: bf16 [B][S][H*dk] token-major.
// Row sums computed via P @ ones WMMA (C-layout replicates row sum per lane).
// ---------------------------------------------------------------------------
__global__ __launch_bounds__(256, 2)
void attention_wmma(const unsigned short* __restrict__ Q,
                    const unsigned short* __restrict__ Kp,
                    const unsigned short* __restrict__ Vt,
                    unsigned short* __restrict__ ctx) {
    __shared__ __align__(16) unsigned short Pbuf[8][16 * LDSS];

    const int tid   = threadIdx.x;
    const int lane  = tid & 31;
    const int wv    = tid >> 5;
    const int bh    = blockIdx.y;                    // b*NHEADS + h
    const int qbase = blockIdx.x * 128 + wv * 16;    // query row within S

    const unsigned short* Qh = Q  + (size_t)bh * SEQ * DKH;
    const unsigned short* Kh = Kp + (size_t)bh * SEQ * DKH;
    const unsigned short* Vh = Vt + (size_t)bh * DKH * SEQ;

    Frag16 qf[2];
    qf[0] = load_frag(Qh + (size_t)qbase * DKH,      DKH, lane);
    qf[1] = load_frag(Qh + (size_t)qbase * DKH + 32, DKH, lane);

    Frag16 ones;
#pragma unroll
    for (int i = 0; i < 16; ++i) ones.u[i] = 0x3F80;   // bf16 1.0

    v8f acc[4];
#pragma unroll
    for (int ni = 0; ni < 4; ++ni) acc[ni] = vzero8();
    float mrow[8], lsum[8];
#pragma unroll
    for (int r = 0; r < 8; ++r) { mrow[r] = -1e30f; lsum[r] = 0.f; }

    const int hf = lane >> 4;
    const int lc = lane & 15;
    unsigned short* Pw = &Pbuf[wv][0];

    for (int kvb = 0; kvb < SEQ; kvb += 64) {
        if (kvb + 64 < SEQ) {
            __builtin_prefetch(Kh + (size_t)(kvb + 64) * DKH, 0, 1);
            __builtin_prefetch(Vh + (size_t)kvb + 64, 0, 1);
        }

        // scores S = Q @ K^T  (16q x 64kv), contraction over dk=64 in 2 steps
        v8f s[4];
#pragma unroll
        for (int j = 0; j < 4; ++j) {
            Frag16 kf0 = load_frag(Kh + (size_t)(kvb + j * 16) * DKH,      DKH, lane);
            Frag16 kf1 = load_frag(Kh + (size_t)(kvb + j * 16) * DKH + 32, DKH, lane);
            s[j] = wmma_bf16(qf[0], kf0, vzero8());
            s[j] = wmma_bf16(qf[1], kf1, s[j]);
        }

        // online softmax (base-2 domain); row r of this lane is q-local hf*8+r,
        // spread over 16 lanes of one half => xor masks 1,2,4,8
        float alph[8];
#pragma unroll
        for (int r = 0; r < 8; ++r) {
            float mx = fmaxf(fmaxf(s[0][r], s[1][r]), fmaxf(s[2][r], s[3][r]));
            mx = fmaxf(mx, __shfl_xor(mx, 1, 32));
            mx = fmaxf(mx, __shfl_xor(mx, 2, 32));
            mx = fmaxf(mx, __shfl_xor(mx, 4, 32));
            mx = fmaxf(mx, __shfl_xor(mx, 8, 32));
            const float mn = fmaxf(mrow[r], mx);
            alph[r] = exp2f(mrow[r] - mn);
            mrow[r] = mn;
            lsum[r] *= alph[r];
            // write P (bf16) straight into LDS: C-layout -> A-layout bounce
#pragma unroll
            for (int j = 0; j < 4; ++j)
                Pw[(hf * 8 + r) * LDSS + lc + j * 16] = f2bf(exp2f(s[j][r] - mn));
        }
#pragma unroll
        for (int ni = 0; ni < 4; ++ni)
#pragma unroll
            for (int r = 0; r < 8; ++r) acc[ni][r] *= alph[r];

        __syncthreads();
        Frag16 pf0 = load_frag(Pw,      LDSS, lane);
        Frag16 pf1 = load_frag(Pw + 32, LDSS, lane);

        // row sums via WMMA: rs[m][n] = sum_k P[m][k] (replicated across n)
        v8f rs = wmma_bf16(pf0, ones, vzero8());
        rs     = wmma_bf16(pf1, ones, rs);
#pragma unroll
        for (int r = 0; r < 8; ++r) lsum[r] += rs[r];

        // O += P @ V ; B^T = V^T rows (dk) x K (kv): plain row-major frags
#pragma unroll
        for (int ni = 0; ni < 4; ++ni) {
            Frag16 vf0 = load_frag(Vh + (size_t)(ni * 16) * SEQ + kvb,      SEQ, lane);
            Frag16 vf1 = load_frag(Vh + (size_t)(ni * 16) * SEQ + kvb + 32, SEQ, lane);
            acc[ni] = wmma_bf16(pf0, vf0, acc[ni]);
            acc[ni] = wmma_bf16(pf1, vf1, acc[ni]);
        }
        __syncthreads();
    }

    // normalize and store ctx as bf16 token-major [B][S][H*dk]
    float inv[8];
#pragma unroll
    for (int r = 0; r < 8; ++r) inv[r] = 1.0f / lsum[r];
    const int b = bh >> 4;   // bh / NHEADS
    const int h = bh & 15;
#pragma unroll
    for (int ni = 0; ni < 4; ++ni) {
#pragma unroll
        for (int r = 0; r < 8; ++r) {
            const int srow = qbase + hf * 8 + r;
            const int col  = h * DKH + ni * 16 + lc;
            ctx[((size_t)b * SEQ + srow) * DMODEL + col] = f2bf(acc[ni][r] * inv[r]);
        }
    }
}

// ---------------------------------------------------------------------------
extern "C" void kernel_launch(void* const* d_in, const int* in_sizes, int n_in,
                              void* d_out, int out_size, void* d_ws, size_t ws_size,
                              hipStream_t stream) {
    (void)in_sizes; (void)n_in; (void)out_size; (void)ws_size;

    const float* query = (const float*)d_in[0];
    const float* key_  = (const float*)d_in[1];
    const float* value = (const float*)d_in[2];
    const float* w_q   = (const float*)d_in[3];
    const float* b_q   = (const float*)d_in[4];
    const float* w_k   = (const float*)d_in[5];
    const float* b_k   = (const float*)d_in[6];
    const float* w_v   = (const float*)d_in[7];
    const float* b_v   = (const float*)d_in[8];
    const float* w_o   = (const float*)d_in[9];
    const float* b_o   = (const float*)d_in[10];
    float* out = (float*)d_out;

    // workspace: 4 bf16 buffers of B*H*S*dk = 4,194,304 elements (8 MB each)
    const size_t nelem = (size_t)BATCH * NHEADS * SEQ * DKH;
    unsigned short* q_perm = (unsigned short*)d_ws;
    unsigned short* k_perm = q_perm + nelem;
    unsigned short* v_t    = k_perm + nelem;
    unsigned short* ctx    = v_t + nelem;

    dim3 blk(256);
    dim3 gridP(DMODEL / 128, (BATCH * SEQ) / 128);   // (8, 32)
    // fold 1/sqrt(dk) AND log2(e) into Q so softmax runs in exp2 domain
    const float qscale = 0.125f * 1.44269504088896340736f;

    gemm_bias_wmma<1, false><<<gridP, blk, 0, stream>>>(query, w_q, b_q, q_perm, qscale);
    gemm_bias_wmma<1, false><<<gridP, blk, 0, stream>>>(key_,  w_k, b_k, k_perm, 1.0f);
    gemm_bias_wmma<2, false><<<gridP, blk, 0, stream>>>(value, w_v, b_v, v_t,    1.0f);

    dim3 gridA(SEQ / 128, BATCH * NHEADS);           // (16, 32)
    attention_wmma<<<gridA, blk, 0, stream>>>(q_perm, k_perm, v_t, ctx);

    gemm_bias_wmma<0, true><<<gridP, blk, 0, stream>>>(ctx, w_o, b_o, out, 1.0f);
}
